// CausalSelfAttention_60808146977382
// MI455X (gfx1250) — compile-verified, matrix operand order checked by reading, not run on hardware
//
#include <hip/hip_runtime.h>
#include <hip/hip_bf16.h>

// Problem constants (reference: B=2, T=2048, C=1024, H=16, D=64)
#define BATCH  2
#define TSEQ   2048
#define CDIM   1024
#define NHEADS 16
#define HDIM   64
#define NQKV   (3 * CDIM)          // 3072
#define MROWS  (BATCH * TSEQ)      // 4096

typedef __bf16 bf16;
typedef __attribute__((ext_vector_type(16))) __bf16 v16bf;
typedef __attribute__((ext_vector_type(8)))  float  v8f;
typedef __attribute__((ext_vector_type(4)))  int    v4i;

#define WMMA_BF16(a, b, c) \
  __builtin_amdgcn_wmma_f32_16x16x32_bf16(false, (a), false, (b), (short)0, (c), false, false)

// ---------------------------------------------------------------------------
// Async global->LDS copies (CDNA5 GLOBAL_LOAD_ASYNC_TO_LDS_B128, ASYNCcnt).
// Probe-confirmed signature: (int4 AS1* src, int4 AS3* dst, imm offset, imm cpol)
#if defined(__has_builtin)
#if __has_builtin(__builtin_amdgcn_global_load_async_to_lds_b128) && \
    __has_builtin(__builtin_amdgcn_s_wait_asynccnt)
#define USE_ASYNC_LDS 1
#endif
#endif

__device__ __forceinline__ void cp16(void* lds_dst, const void* gsrc) {
#ifdef USE_ASYNC_LDS
  __builtin_amdgcn_global_load_async_to_lds_b128(
      (__attribute__((address_space(1))) v4i*)(gsrc),
      (__attribute__((address_space(3))) v4i*)(lds_dst),
      0, 0);
#else
  *(uint4*)lds_dst = *(const uint4*)gsrc;
#endif
}
__device__ __forceinline__ void cp_wait() {
#ifdef USE_ASYNC_LDS
  __builtin_amdgcn_s_wait_asynccnt(0);
#endif
}

// ---------------------------------------------------------------------------
// Fragment helpers (ISA 7.12.2 layouts, wave32). All hot-loop loads contiguous.
// A fragment: 16x32 bf16 row-major, ld = leading dim. Per lane: two 16B loads.
__device__ __forceinline__ v16bf frag_a(const bf16* p, int ld) {
  int lane = threadIdx.x & 31;
  int row  = lane & 15;
  int kb   = (lane >> 4) << 3;          // 0 or 8
  const bf16* r = p + row * ld;
  v16bf a;
#pragma unroll
  for (int h = 0; h < 2; ++h)
#pragma unroll
    for (int j = 0; j < 4; ++j) {
      int k = h * 16 + kb + 2 * j;
      a[h * 8 + 2 * j]     = r[k];
      a[h * 8 + 2 * j + 1] = r[k + 1];
    }
  return a;
}

// B fragment from transposed view: B[k][n] = src[n*ld + k]. Contiguous in k.
__device__ __forceinline__ v16bf frag_bT(const bf16* src, int ld) {
  int lane = threadIdx.x & 31;
  int col  = lane & 15;
  int kb   = (lane >> 4) << 3;
  const bf16* r = src + (size_t)col * ld;
  v16bf b;
#pragma unroll
  for (int h = 0; h < 2; ++h)
#pragma unroll
    for (int j = 0; j < 4; ++j) {
      int k = h * 16 + kb + 2 * j;
      b[h * 8 + 2 * j]     = r[k];
      b[h * 8 + 2 * j + 1] = r[k + 1];
    }
  return b;
}

// B fragment from pre-swizzled fragment-major weights: lane's 16 values are
// 32 contiguous bytes -> two global_load_b128, no packing VALU.
__device__ __forceinline__ v16bf frag_b_packed(const bf16* base) {
  int lane = threadIdx.x & 31;
  return *(const v16bf*)(base + lane * 16);
}

// C fragment store: VGPR r -> row r (lanes 0-15) / r+8 (lanes 16-31), col = lane&15
__device__ __forceinline__ void store_c(float* dst, int ld, v8f c) {
  int lane = threadIdx.x & 31;
  int col  = lane & 15;
  int rb   = (lane >> 4) << 3;
#pragma unroll
  for (int r = 0; r < 8; ++r)
    dst[(size_t)(r + rb) * ld + col] = c[r];
}

// ---------------------------------------------------------------------------
// fp32 -> bf16 (row-major kept; used for x)
__global__ void f32_to_bf16_kernel(const float* __restrict__ src,
                                   bf16* __restrict__ dst, int n) {
  for (int i = blockIdx.x * blockDim.x + threadIdx.x; i < n;
       i += gridDim.x * blockDim.x)
    dst[i] = (bf16)src[i];
}

// Pack fp32 weights [K x N] into WMMA-B fragment-major bf16:
// out[((kt*(N/16)+nt)*32 + lane)*16 + e] = B-fragment element e of lane.
// One-time gather; makes every hot-loop B load contiguous.
__global__ void pack_w_kernel(const float* __restrict__ W,
                              bf16* __restrict__ out, int K, int N) {
  int NT = N >> 4;
  int total = (K >> 5) * NT * 32;
  for (int g = blockIdx.x * blockDim.x + threadIdx.x; g < total;
       g += gridDim.x * blockDim.x) {
    int lane = g & 31;
    int tile = g >> 5;
    int nt = tile % NT, kt = tile / NT;
    int col = lane & 15, kb = (lane >> 4) << 3;
    bf16* o = out + (size_t)g * 16;
#pragma unroll
    for (int h = 0; h < 2; ++h)
#pragma unroll
      for (int j = 0; j < 4; ++j) {
        int k = h * 16 + kb + 2 * j;
        o[h * 8 + 2 * j] =
            (bf16)W[(size_t)(kt * 32 + k) * N + nt * 16 + col];
        o[h * 8 + 2 * j + 1] =
            (bf16)W[(size_t)(kt * 32 + k + 1) * N + nt * 16 + col];
      }
  }
}

// ---------------------------------------------------------------------------
// QKV GEMM (4096x1024 @ 1024x3072) + fused RoPE/scale/head-transpose epilogue.
// 256 thr = 8 waves, 64x64 tile, K-step 64. V is written TRANSPOSED ([B,H,D,T])
// so the attention PV B-fragments become contiguous LDS loads.
__global__ void qkv_rope_kernel(const bf16* __restrict__ xb,
                                const bf16* __restrict__ wpk,   // packed Wqkv
                                bf16* __restrict__ qb,
                                bf16* __restrict__ kbuf,
                                bf16* __restrict__ vtb) {
  __shared__ bf16  sA[64 * 64];
  __shared__ float sC[64 * 64];
  int m0 = blockIdx.x * 64, n0 = blockIdx.y * 64;
  int w = threadIdx.x >> 5;
  int m16 = w & 3, np = w >> 2;
  const int NT = NQKV >> 4;          // 192 n-tiles
  v8f acc[2];
  acc[0] = 0; acc[1] = 0;

  for (int k0 = 0; k0 < CDIM; k0 += 64) {
    __syncthreads();
#pragma unroll
    for (int c = 0; c < 2; ++c) {
      int idx = threadIdx.x + c * 256;       // 512 x 16B = 64x64 bf16
      int r = idx >> 3, c8 = (idx & 7) * 8;
      cp16(sA + r * 64 + c8, xb + (size_t)(m0 + r) * CDIM + k0 + c8);
    }
    cp_wait();
    __syncthreads();
#pragma unroll
    for (int kk = 0; kk < 2; ++kk) {
      v16bf a = frag_a(sA + m16 * 16 * 64 + kk * 32, 64);
      int kt = (k0 >> 5) + kk;
#pragma unroll
      for (int t = 0; t < 2; ++t) {
        int nt = (n0 >> 4) + np * 2 + t;
        v16bf b = frag_b_packed(wpk + ((size_t)kt * NT + nt) * 512);
        acc[t] = WMMA_BF16(a, b, acc[t]);
      }
    }
  }
  __syncthreads();
#pragma unroll
  for (int t = 0; t < 2; ++t)
    store_c(sC + (m16 * 16) * 64 + (np * 2 + t) * 16, 64, acc[t]);
  __syncthreads();

  // Epilogue: RoPE for q,k (1/sqrt(D) folded into q), v -> transposed layout
  int s = n0 / CDIM;                 // 0=q 1=k 2=v
  int h = (n0 % CDIM) / HDIM;
  int i = threadIdx.x;
  int row = i >> 2;
  int m  = m0 + row;
  int bb = m / TSEQ, tt = m % TSEQ;
  size_t rbase = ((size_t)(bb * NHEADS + h) * TSEQ + tt) * HDIM;   // q/k rows
  size_t vbase = (size_t)(bb * NHEADS + h) * HDIM;                 // v^T rows
#pragma unroll
  for (int e = 0; e < 16; ++e) {
    int d = (i & 3) * 16 + e;
    float v = sC[row * 64 + d];
    if (s < 2) {
      float rot = (d < 32) ? -sC[row * 64 + d + 32] : sC[row * 64 + d - 32];
      // inv_freq = 10000^(-(d&31)/32) = exp(-(ln 1e4)/32 * (d&31))
      float ang = (float)tt * __expf(-0.28782314f * (float)(d & 31));
      float outv = v * __cosf(ang) + rot * __sinf(ang);
      if (s == 0) {
        qb[rbase + d] = (bf16)(outv * 0.125f);     // 1/sqrt(64)
      } else {
        kbuf[rbase + d] = (bf16)outv;
      }
    } else {
      vtb[(vbase + d) * TSEQ + tt] = (bf16)v;      // V^T [B,H,D,T]
    }
  }
}

// ---------------------------------------------------------------------------
// Flash attention: grid (B*H, T/64). 128 thr = 4 waves; wave owns a 16-query
// strip. K (row-major) and V^T tiles staged via async-to-LDS; QK^T and PV both
// consume contiguous ds_load_b128 fragments; online softmax via shfl_xor over
// the 16-lane row halves of the C layout.
__global__ void attn_kernel(const bf16* __restrict__ qb,
                            const bf16* __restrict__ kb,
                            const bf16* __restrict__ vtb,
                            bf16* __restrict__ ob) {
  __shared__ bf16 sK [64 * HDIM];       // [key][d]
  __shared__ bf16 sVt[HDIM * 64];       // [d][key]
  __shared__ bf16 sP [4 * 16 * HDIM];   // per-wave P strips
  int bh = blockIdx.x;               // b*16 + h
  int qt = blockIdx.y;               // 64-query tile
  int w    = threadIdx.x >> 5;
  int lane = threadIdx.x & 31;
  int col  = lane & 15;
  int half = lane >> 4;

  size_t headbase = (size_t)bh * TSEQ * HDIM;   // same for q/k and v^T
  const bf16* Q = qb + headbase + (size_t)(qt * 64 + w * 16) * HDIM;
  v16bf qa0 = frag_a(Q, HDIM);       // d = 0..31
  v16bf qa1 = frag_a(Q + 32, HDIM);  // d = 32..63

  v8f o[4];
  o[0] = 0; o[1] = 0; o[2] = 0; o[3] = 0;
  float mrow[8], lrow[8];
#pragma unroll
  for (int r = 0; r < 8; ++r) { mrow[r] = -1e30f; lrow[r] = 0.f; }

  for (int j = 0; j <= qt; ++j) {
    __syncthreads();  // previous iteration done with sK/sVt/sP
    const bf16* ktile = kb  + headbase + (size_t)j * 64 * HDIM;  // 8KB contig
    const bf16* vtile = vtb + headbase + (size_t)j * 64;         // 64 strided rows
    for (int i = threadIdx.x; i < 512; i += 128) {
      int r = i >> 3, c8 = (i & 7) * 8;
      cp16(sK + i * 8, ktile + i * 8);
      cp16(sVt + r * 64 + c8, vtile + (size_t)r * TSEQ + c8);
    }
    if (j < qt) {  // gfx1250 global_prefetch of next tiles
      __builtin_prefetch(ktile + 64 * HDIM, 0, 3);
      __builtin_prefetch(vtile + 64, 0, 3);
    }
    cp_wait();
    __syncthreads();

    // S = Q K^T (16x64 per wave; 2 WMMAs per 16x16 subtile over d=64)
    v8f s[4];
    s[0] = 0; s[1] = 0; s[2] = 0; s[3] = 0;
#pragma unroll
    for (int n = 0; n < 4; ++n) {
      s[n] = WMMA_BF16(qa0, frag_bT(sK + n * 16 * HDIM, HDIM), s[n]);
      s[n] = WMMA_BF16(qa1, frag_bT(sK + n * 16 * HDIM + 32, HDIM), s[n]);
    }

    if (j == qt) {  // causal mask on the diagonal tile
#pragma unroll
      for (int n = 0; n < 4; ++n)
#pragma unroll
        for (int r = 0; r < 8; ++r) {
          int rg = w * 16 + r + half * 8;
          int cg = n * 16 + col;
          if (cg > rg) s[n][r] = -1e30f;
        }
    }

    // Online softmax per fragment row (16-lane half = one row group)
#pragma unroll
    for (int r = 0; r < 8; ++r) {
      float rm = fmaxf(fmaxf(s[0][r], s[1][r]), fmaxf(s[2][r], s[3][r]));
#pragma unroll
      for (int off = 8; off >= 1; off >>= 1)
        rm = fmaxf(rm, __shfl_xor(rm, off, 32));
      float mnew = fmaxf(mrow[r], rm);
      float scl  = __expf(mrow[r] - mnew);
      float rs = 0.f;
#pragma unroll
      for (int n = 0; n < 4; ++n) {
        float p = __expf(s[n][r] - mnew);
        s[n][r] = p;
        rs += p;
      }
#pragma unroll
      for (int off = 8; off >= 1; off >>= 1)
        rs += __shfl_xor(rs, off, 32);
      lrow[r] = lrow[r] * scl + rs;
      mrow[r] = mnew;
#pragma unroll
      for (int n = 0; n < 4; ++n) o[n][r] *= scl;
    }

    // P -> LDS (A-fragment reload), then O += P @ V via V^T fragments
    bf16* P = sP + w * 16 * HDIM;
#pragma unroll
    for (int n = 0; n < 4; ++n)
#pragma unroll
      for (int r = 0; r < 8; ++r)
        P[(r + half * 8) * HDIM + n * 16 + col] = (bf16)s[n][r];
    __syncthreads();

    v16bf pa0 = frag_a(P, HDIM);       // keys 0..31
    v16bf pa1 = frag_a(P + 32, HDIM);  // keys 32..63
#pragma unroll
    for (int n = 0; n < 4; ++n) {
      o[n] = WMMA_BF16(pa0, frag_bT(sVt + n * 16 * 64, 64), o[n]);
      o[n] = WMMA_BF16(pa1, frag_bT(sVt + n * 16 * 64 + 32, 64), o[n]);
    }
  }

  // O /= l, write to [B, T, C] (heads re-concatenated) as bf16
  int b = bh >> 4, h = bh & 15;
  int t0 = qt * 64 + w * 16;
#pragma unroll
  for (int r = 0; r < 8; ++r) {
    float inv = 1.0f / lrow[r];
    int tt = t0 + r + half * 8;
    size_t rowbase = ((size_t)(b * TSEQ + tt)) * CDIM + h * HDIM;
#pragma unroll
    for (int n = 0; n < 4; ++n)
      ob[rowbase + n * 16 + col] = (bf16)(o[n][r] * inv);
  }
}

// ---------------------------------------------------------------------------
// Output projection GEMM: bf16 A (row-major) x packed-B, fp32 out. K-step 64.
__global__ void gemm_bf16_kernel(const bf16* __restrict__ A,
                                 const bf16* __restrict__ Bpk,
                                 float* __restrict__ Cm, int K, int N) {
  __shared__ bf16 sA[64 * 64];
  int m0 = blockIdx.x * 64, n0 = blockIdx.y * 64;
  int w = threadIdx.x >> 5;
  int m16 = w & 3, np = w >> 2;
  int NT = N >> 4;
  v8f acc[2];
  acc[0] = 0; acc[1] = 0;
  for (int k0 = 0; k0 < K; k0 += 64) {
    __syncthreads();
#pragma unroll
    for (int c = 0; c < 2; ++c) {
      int idx = threadIdx.x + c * 256;
      int r = idx >> 3, c8 = (idx & 7) * 8;
      cp16(sA + r * 64 + c8, A + (size_t)(m0 + r) * K + k0 + c8);
    }
    cp_wait();
    __syncthreads();
#pragma unroll
    for (int kk = 0; kk < 2; ++kk) {
      v16bf a = frag_a(sA + m16 * 16 * 64 + kk * 32, 64);
      int kt = (k0 >> 5) + kk;
#pragma unroll
      for (int t = 0; t < 2; ++t) {
        int nt = (n0 >> 4) + np * 2 + t;
        v16bf b = frag_b_packed(Bpk + ((size_t)kt * NT + nt) * 512);
        acc[t] = WMMA_BF16(a, b, acc[t]);
      }
    }
  }
#pragma unroll
  for (int t = 0; t < 2; ++t)
    store_c(Cm + (size_t)(m0 + m16 * 16) * N + n0 + (np * 2 + t) * 16, N, acc[t]);
}

// ---------------------------------------------------------------------------
extern "C" void kernel_launch(void* const* d_in, const int* in_sizes, int n_in,
                              void* d_out, int out_size, void* d_ws, size_t ws_size,
                              hipStream_t stream) {
  (void)in_sizes; (void)n_in; (void)out_size; (void)ws_size;
  const float* x    = (const float*)d_in[0];
  const float* Wqkv = (const float*)d_in[1];
  const float* Wout = (const float*)d_in[2];
  float* out = (float*)d_out;

  char* ws = (char*)d_ws;
  // Workspace map (48 MiB total)
  bf16* xb    = (bf16*)(ws);                        //  8 MiB: x bf16 row-major
  bf16* wqkvp = (bf16*)(ws + (size_t)( 8u << 20));  //  6 MiB: packed Wqkv
  bf16* woutp = (bf16*)(ws + (size_t)(14u << 20));  //  2 MiB: packed Wout
  bf16* qb    = (bf16*)(ws + (size_t)(16u << 20));  //  8 MiB: q [B,H,T,D] RoPE'd*1/8
  bf16* kbuf  = (bf16*)(ws + (size_t)(24u << 20));  //  8 MiB: k [B,H,T,D] RoPE'd
  bf16* vtb   = (bf16*)(ws + (size_t)(32u << 20));  //  8 MiB: v^T [B,H,D,T]
  bf16* ab    = (bf16*)(ws + (size_t)(40u << 20));  //  8 MiB: attn out [B,T,C]

  f32_to_bf16_kernel<<<1024, 256, 0, stream>>>(x, xb, MROWS * CDIM);
  pack_w_kernel<<<768, 256, 0, stream>>>(Wqkv, wqkvp, CDIM, NQKV);
  pack_w_kernel<<<256, 256, 0, stream>>>(Wout, woutp, CDIM, CDIM);

  qkv_rope_kernel<<<dim3(MROWS / 64, NQKV / 64), 256, 0, stream>>>(
      xb, wqkvp, qb, kbuf, vtb);

  attn_kernel<<<dim3(BATCH * NHEADS, TSEQ / 64), 128, 0, stream>>>(
      qb, kbuf, vtb, ab);

  gemm_bf16_kernel<<<dim3(MROWS / 64, CDIM / 64), 256, 0, stream>>>(
      ab, woutp, out, CDIM, CDIM);
}
